// DenseBlock_33921651704539
// MI455X (gfx1250) — compile-verified, matrix operand order checked by reading, not run on hardware
//
#include <hip/hip_runtime.h>
#include <hip/hip_bf16.h>
#include <stdint.h>

#define NLAYERS 6
#define CIN 256
#define GR 32
#define CF 448           // 256 + 6*32
#define HH 56
#define WW 56
#define NB 32
#define HW (HH*WW)

typedef __attribute__((ext_vector_type(16))) __bf16 v16bf;
typedef __attribute__((ext_vector_type(8)))  __bf16 v8bf;
typedef __attribute__((ext_vector_type(8)))  float  v8f;
typedef __attribute__((ext_vector_type(4)))  float  v4f;

__device__ __forceinline__ unsigned short f2bf(float f) {
    unsigned u = __float_as_uint(f);
    unsigned r = (u + 0x7FFFu + ((u >> 16) & 1u)) >> 16;   // round-to-nearest-even
    return (unsigned short)r;
}
__device__ __forceinline__ float bf2f(unsigned short h) {
    return __uint_as_float(((unsigned)h) << 16);
}

// ---------------------------------------------------------------------------
// 1) Copy x (NCHW fp32) into d_out channels [0, CIN) of the NCHW [NB,CF,H,W]
// ---------------------------------------------------------------------------
__global__ void copy_x_kernel(const v4f* __restrict__ x, float* __restrict__ out) {
    long i = (long)blockIdx.x * blockDim.x + threadIdx.x;
    long total = (long)NB * CIN * HW / 4;
    if (i >= total) return;
    v4f v = x[i];
    long e = i * 4;
    int p = (int)(e % HW);
    long t = e / HW;
    int ch = (int)(t % CIN);
    int n  = (int)(t / CIN);
    __builtin_nontemporal_store(v, (v4f*)&out[((long)n * CF + ch) * HW + p]);
}

// ---------------------------------------------------------------------------
// 2) Transpose x NCHW fp32 -> feat NHWC bf16 (channel stride CF)
// ---------------------------------------------------------------------------
__global__ void x_to_feat_kernel(const float* __restrict__ x, unsigned short* __restrict__ feat) {
    __shared__ float tile[64][33];
    int ch0 = blockIdx.y * 64;
    long fp0 = (long)blockIdx.x * 32;      // HW % 32 == 0: never crosses image
    int n  = (int)(fp0 / HW);
    int p0 = (int)(fp0 - (long)n * HW);

    int lp = threadIdx.x & 31;             // pixel
    int lc = threadIdx.x >> 5;             // 0..7
#pragma unroll
    for (int r = 0; r < 8; ++r) {
        int ch = lc + r * 8;
        tile[ch][lp] = x[((long)n * CIN + ch0 + ch) * HW + p0 + lp];
    }
    __syncthreads();
    int sp = threadIdx.x >> 3;             // pixel 0..31
    int g  = threadIdx.x & 7;              // channel group of 8
    unsigned short tmp[8] __attribute__((aligned(16)));
#pragma unroll
    for (int j = 0; j < 8; ++j) tmp[j] = f2bf(tile[g * 8 + j][sp]);
    *(uint4*)&feat[((long)n * HW + p0 + sp) * CF + ch0 + g * 8] = *(uint4*)tmp;
}

// ---------------------------------------------------------------------------
// 3) BN scale/shift precompute (per layer)
// ---------------------------------------------------------------------------
__global__ void bn_prep_kernel(const float* __restrict__ w, const float* __restrict__ b,
                               const float* __restrict__ m, const float* __restrict__ v,
                               float* __restrict__ scale, float* __restrict__ shift, int c) {
    int i = blockIdx.x * blockDim.x + threadIdx.x;
    if (i >= c) return;
    float inv = rsqrtf(v[i] + 1e-5f);
    float s = w[i] * inv;
    scale[i] = s;
    shift[i] = b[i] - m[i] * s;
}

// ---------------------------------------------------------------------------
// 4) Weight repack: OIHW fp32 -> [o][K] bf16, K ordered (kh, kw, ch)
// ---------------------------------------------------------------------------
__global__ void wt_prep_kernel(const float* __restrict__ wsrc, unsigned short* __restrict__ wt, int c) {
    int Ktot = 9 * c;
    long i = (long)blockIdx.x * blockDim.x + threadIdx.x;
    if (i >= (long)GR * Ktot) return;
    int k = (int)(i % Ktot);
    int o = (int)(i / Ktot);
    int tap = k / c;
    int ch  = k - tap * c;
    int kh = tap / 3, kw = tap - kh * 3;
    wt[i] = f2bf(wsrc[(((long)o * c + ch) * 3 + kh) * 3 + kw]);
}

// ---------------------------------------------------------------------------
// 5) Fused BN+ReLU + 3x3 conv as implicit GEMM with bf16 WMMA.
//    Block = one (n, y) image row: 64 x-positions x 32 out-channels.
//    8 waves = (mt 0..3) x (nt 0..1); each wave one 16x16 f32 C tile.
//    Double-buffered LDS, one barrier per 32-channel chunk, next-chunk
//    global loads issued before the barrier to overlap with WMMA.
// ---------------------------------------------------------------------------
#define SROW 40    // stage row stride (halfwords): 32 + 8 pad, 80B -> conflict-free b128
#define WROW 104   // weight row stride (halfwords): 96 + 8 pad, 208B

// BN+ReLU on 8 bf16 channels packed in a uint4, repacked to bf16 with
// +0x8000 round + v_perm_b32 high-half pack (3 VALU per pair).
__device__ __forceinline__ void bn_relu_pack(unsigned short* dst, uint4 raw,
                                             const float* sc, const float* sh,
                                             int ch, bool valid) {
    unsigned ru[4] = {raw.x, raw.y, raw.z, raw.w};
    unsigned pv[4];
#pragma unroll
    for (int q = 0; q < 4; ++q) {
        int j = 2 * q;
        float f0 = __uint_as_float(ru[q] << 16);
        float f1 = __uint_as_float(ru[q] & 0xffff0000u);
        f0 = fmaxf(f0 * sc[ch + j]     + sh[ch + j],     0.f);
        f1 = fmaxf(f1 * sc[ch + j + 1] + sh[ch + j + 1], 0.f);
        unsigned u0 = __float_as_uint(f0) + 0x8000u;
        unsigned u1 = __float_as_uint(f1) + 0x8000u;
        unsigned p  = __builtin_amdgcn_perm(u1, u0, 0x07060302u); // [u1.hi16|u0.hi16]
        pv[q] = valid ? p : 0u;
    }
    uint4 o; o.x = pv[0]; o.y = pv[1]; o.z = pv[2]; o.w = pv[3];
    *(uint4*)dst = o;
}

__global__ __launch_bounds__(256) void dense_conv_kernel(
    const unsigned short* feat,            // NHWC bf16, stride CF (aliases featw)
    const unsigned short* __restrict__ wt, // [GR][9*c] bf16
    const float* __restrict__ scale,
    const float* __restrict__ shift,
    float* __restrict__ out,               // NCHW [NB,CF,HW] fp32
    unsigned short* featw,                 // NHWC bf16 (write channels >= c)
    int c, int co0)
{
    __shared__ __align__(16) unsigned short stage[2][66 * SROW];
    __shared__ __align__(16) unsigned short wlds[2][32 * WROW];
    __shared__ float sSc[416];
    __shared__ float sSh[416];

    const int Ktot = 9 * c;
    const int blk  = blockIdx.x;           // 0 .. NB*HH-1
    const int n    = blk / HH;
    const int y    = blk - n * HH;
    const int tid  = threadIdx.x;

    // preload BN params to LDS
    for (int i = tid; i < c; i += 256) { sSc[i] = scale[i]; sSh[i] = shift[i]; }

    // ---- static per-thread staging descriptors (all shifts/masks) ----
    const int  p0   = tid >> 2;            // activation pixel 0..63
    const int  g0   = tid & 3;             // 8-channel group
    const bool hasA1 = (tid < 8);          // extra pixels 64,65
    const int  p1   = 64 + (tid >> 2);
    const int  w_kw = tid >> 7;            // weight task 0: kw 0..1
    const int  w_o  = (tid >> 2) & 31;
    const int  w_g  = tid & 3;
    const bool hasW1 = (tid < 128);        // weight task 1: kw=2

    // ---- wave / fragment mapping ----
    const int lane  = tid & 31;
    const int wv    = tid >> 5;
    const int mt    = wv >> 1;             // x tile
    const int nt    = wv & 1;              // out-channel tile
    const int x0    = mt * 16;
    const int mrow  = lane & 15;
    const int kbase = (lane < 16) ? 0 : 8; // 16-bit A/B fragment K base per lane half

    v8f acc = {0.f, 0.f, 0.f, 0.f, 0.f, 0.f, 0.f, 0.f};
    const uint4 zero4 = {0u, 0u, 0u, 0u};

    __syncthreads();                       // sSc/sSh ready

    int buf = 0;
    for (int kh = 0; kh < 3; ++kh) {
        const int  ys    = y + kh - 1;
        const bool rowOK = (ys >= 0) && (ys < HH);
        const unsigned short* frow = feat + ((long)n * HW + (long)ys * WW) * CF;

        const int  xi0 = p0 - 1;
        const int  xi1 = p1 - 1;
        const bool v0  = rowOK && (xi0 >= 0) && (xi0 < WW);
        const bool v1  = hasA1 && rowOK && (xi1 >= 0) && (xi1 < WW);
        const unsigned short* a0src = frow + (long)xi0 * CF + g0 * 8;
        const unsigned short* a1src = frow + (long)xi1 * CF + g0 * 8;
        const unsigned short* w0src = wt + (long)w_o * Ktot + (kh * 3 + w_kw) * c + w_g * 8;
        const unsigned short* w1src = wt + (long)w_o * Ktot + (kh * 3 + 2)    * c + w_g * 8;

        // preload chunk cc = 0
        uint4 aReg0 = v0    ? *(const uint4*)a0src : zero4;
        uint4 aReg1 = v1    ? *(const uint4*)a1src : zero4;
        uint4 wReg0 =         *(const uint4*)w0src;
        uint4 wReg1 = hasW1 ? *(const uint4*)w1src : zero4;

        for (int cc = 0; cc < c; cc += 32) {
            // ---- transform + LDS stores (current chunk, buffer `buf`) ----
            bn_relu_pack(&stage[buf][p0 * SROW + g0 * 8], aReg0, sSc, sSh, cc + g0 * 8, v0);
            if (hasA1)
                bn_relu_pack(&stage[buf][p1 * SROW + g0 * 8], aReg1, sSc, sSh, cc + g0 * 8, v1);
            *(uint4*)&wlds[buf][w_o * WROW + w_kw * 32 + w_g * 8] = wReg0;
            if (hasW1)
                *(uint4*)&wlds[buf][w_o * WROW + 64 + w_g * 8] = wReg1;

            // ---- issue next chunk's global loads (fly during barrier+WMMA) ----
            if (cc + 32 < c) {
                const int nc = cc + 32;
                aReg0 = v0    ? *(const uint4*)(a0src + nc) : zero4;
                aReg1 = v1    ? *(const uint4*)(a1src + nc) : zero4;
                wReg0 =         *(const uint4*)(w0src + nc);
                wReg1 = hasW1 ? *(const uint4*)(w1src + nc) : zero4;
            }

            __syncthreads();

            // ---- 3 WMMAs on buffer `buf` ----
#pragma unroll
            for (int kw = 0; kw < 3; ++kw) {
                const v8bf a0 = *(const v8bf*)&stage[buf][(x0 + mrow + kw) * SROW + kbase];
                const v8bf a1 = *(const v8bf*)&stage[buf][(x0 + mrow + kw) * SROW + kbase + 16];
                const v8bf b0 = *(const v8bf*)&wlds[buf][(nt * 16 + mrow) * WROW + kw * 32 + kbase];
                const v8bf b1 = *(const v8bf*)&wlds[buf][(nt * 16 + mrow) * WROW + kw * 32 + kbase + 16];
                v16bf a, b;
#pragma unroll
                for (int j = 0; j < 8; ++j) {
                    a[j] = a0[j]; a[j + 8] = a1[j];
                    b[j] = b0[j]; b[j + 8] = b1[j];
                }
                acc = __builtin_amdgcn_wmma_f32_16x16x32_bf16(
                    false, a, false, b, (short)0, acc, false, false);
            }
            buf ^= 1;
        }
    }

    // ---- epilogue: C layout 16x16 f32 -> NCHW fp32 out (NT) + NHWC bf16 feat ----
    const int nloc = nt * 16 + (lane & 15);      // N = out channel within GROWTH
    const int co   = co0 + nloc;
    const int mhi  = (lane >> 4) << 3;           // lanes 16-31 hold M = r+8
#pragma unroll
    for (int r = 0; r < 8; ++r) {
        int x = x0 + r + mhi;
        if (x < WW) {
            float vv = acc[r];
            __builtin_nontemporal_store(vv, &out[((long)n * CF + co) * HW + (long)y * WW + x]);
            featw[((long)n * HW + (long)y * WW + x) * CF + co] = f2bf(vv);
        }
    }
}

// ---------------------------------------------------------------------------
// Host launch
// ---------------------------------------------------------------------------
extern "C" void kernel_launch(void* const* d_in, const int* in_sizes, int n_in,
                              void* d_out, int out_size, void* d_ws, size_t ws_size,
                              hipStream_t stream) {
    (void)in_sizes; (void)n_in; (void)out_size; (void)ws_size;
    const float* x = (const float*)d_in[0];
    const float *bnw[NLAYERS], *bnb[NLAYERS], *bnm[NLAYERS], *bnv[NLAYERS], *ck[NLAYERS];
    for (int i = 0; i < NLAYERS; ++i) {
        bnw[i] = (const float*)d_in[1 + i];
        bnb[i] = (const float*)d_in[7 + i];
        bnm[i] = (const float*)d_in[13 + i];
        bnv[i] = (const float*)d_in[19 + i];
        ck[i]  = (const float*)d_in[25 + i];
    }
    float* out = (float*)d_out;
    char* ws = (char*)d_ws;

    // Workspace layout
    size_t off = 0;
    unsigned short* feat = (unsigned short*)(ws + off);
    off += (size_t)NB * HW * CF * 2;                      // ~85.8 MB NHWC bf16 feature map
    off = (off + 255) & ~(size_t)255;
    unsigned short* wtp[NLAYERS];
    float *scl[NLAYERS], *shf[NLAYERS];
    for (int i = 0; i < NLAYERS; ++i) {
        int c = CIN + i * GR;
        wtp[i] = (unsigned short*)(ws + off);
        off += (size_t)GR * 9 * c * 2;
        off = (off + 255) & ~(size_t)255;
    }
    for (int i = 0; i < NLAYERS; ++i) {
        int c = CIN + i * GR;
        scl[i] = (float*)(ws + off); off += (size_t)c * 4; off = (off + 255) & ~(size_t)255;
        shf[i] = (float*)(ws + off); off += (size_t)c * 4; off = (off + 255) & ~(size_t)255;
    }

    // 1) x -> d_out channels [0,256)
    {
        long total = (long)NB * CIN * HW / 4;
        copy_x_kernel<<<(int)((total + 255) / 256), 256, 0, stream>>>((const v4f*)x, out);
    }
    // 2) x -> feat NHWC bf16
    {
        dim3 g(NB * HW / 32, CIN / 64);
        x_to_feat_kernel<<<g, 256, 0, stream>>>(x, feat);
    }
    // 3) per-layer BN params + weight repack
    for (int i = 0; i < NLAYERS; ++i) {
        int c = CIN + i * GR;
        bn_prep_kernel<<<(c + 255) / 256, 256, 0, stream>>>(bnw[i], bnb[i], bnm[i], bnv[i],
                                                            scl[i], shf[i], c);
        long wtotal = (long)GR * 9 * c;
        wt_prep_kernel<<<(int)((wtotal + 255) / 256), 256, 0, stream>>>(ck[i], wtp[i], c);
    }
    // 4) fused BN+ReLU+conv per layer (stream order serializes layers)
    for (int i = 0; i < NLAYERS; ++i) {
        int c = CIN + i * GR;
        dense_conv_kernel<<<NB * HH, 256, 0, stream>>>(feat, wtp[i], scl[i], shf[i],
                                                       out, feat, c, CIN + i * GR);
    }
}
